// DepthlessTransformer_51058571215163
// MI455X (gfx1250) — matrix operand order repaired and checked by reading, not compiled, not run on hardware
//
#include <hip/hip_runtime.h>
#include <hip/hip_bf16.h>

// ---------------------------------------------------------------------------
// DepthlessTransformer forward for MI455X (gfx1250, wave32, WMMA).
// All GEMMs (95% of the ~0.5 TFLOP) run on v_wmma_f32_16x16x32_f16 with f32
// accumulation; softmax / RMSNorm / exact-GELU / sigmoid stay in fp32 VALU.
// GEMM wave tile is 32x64 (2 A-frags x 4 B-frags -> 8 WMMA per K-step) for
// register-level operand reuse: 0.75 fragment-loads per WMMA.
// ---------------------------------------------------------------------------

typedef __attribute__((ext_vector_type(16))) _Float16 v16h;
typedef __attribute__((ext_vector_type(8)))  _Float16 v8h;
typedef __attribute__((ext_vector_type(8)))  float    v8f;

#define EPSR 1.1920929e-07f   // torch float32 eps used by RMSNorm in reference

__device__ __forceinline__ v8f vzero8() {
  v8f z = {0.f,0.f,0.f,0.f,0.f,0.f,0.f,0.f};
  return z;
}

// A fragment (16x32 f16, ISA 7.12.2): lane r=lane&15 holds row r; lane-half g
// holds K runs [k0+8g .. +7] and [k0+16+8g .. +7]  (two 16B contiguous loads).
__device__ __forceinline__ v16h frag_a_g(const _Float16* A, long row0, int lda, int k0) {
  int lane = threadIdx.x & 31;
  int r = lane & 15, g = lane >> 4;
  const _Float16* p = A + (row0 + r) * (long)lda + k0 + g * 8;
  v8h lo = *(const v8h*)(p);
  v8h hi = *(const v8h*)(p + 16);
  return __builtin_shufflevector(lo, hi, 0,1,2,3,4,5,6,7,8,9,10,11,12,13,14,15);
}

// B fragment (32x16 f16) loaded from B^T stored row-major (N x K): lane holds
// column n0+(lane&15), K runs [k0+8g .. +7] and [k0+16+8g .. +7].
__device__ __forceinline__ v16h frag_b_g(const _Float16* BT, long n0, int ldb, int k0) {
  int lane = threadIdx.x & 31;
  int c = lane & 15, g = lane >> 4;
  const _Float16* p = BT + (n0 + c) * (long)ldb + k0 + g * 8;
  v8h lo = *(const v8h*)(p);
  v8h hi = *(const v8h*)(p + 16);
  return __builtin_shufflevector(lo, hi, 0,1,2,3,4,5,6,7,8,9,10,11,12,13,14,15);
}

// A fragment built on the fly from fp32 softmax probabilities in LDS.
__device__ __forceinline__ v16h frag_a_lds_f32(const float* S, int ldS, int j0) {
  int lane = threadIdx.x & 31;
  int r = lane & 15, g = lane >> 4;
  const float* p = S + r * ldS + j0 + g * 8;
  v16h out;
#pragma unroll
  for (int i = 0; i < 8; ++i) { out[i] = (_Float16)p[i]; out[8 + i] = (_Float16)p[16 + i]; }
  return out;
}

// ---------------------------------------------------------------------------
// Generic WMMA GEMM: C(MxN,f32-acc) = A(MxK,f16) * B(KxN) (+bias), B given as
// B^T (NxK,f16).  Per-row-block weight selection supports the L-ensemble.
// EPI: 0 = store f32, 1 = store f16, 2 = per-head(64-col) RMSNorm*wnorm -> f16
// Wave computes a 32x64 tile (8 wmma accumulators). 8 waves / workgroup.
// Requires M % 32 == 0, N % 64 == 0, K % 32 == 0 (all call sites satisfy).
// ---------------------------------------------------------------------------
template<int EPI, int HASBIAS>
__global__ void k_gemm(const _Float16* __restrict__ A, int lda,
                       const _Float16* __restrict__ BT, int ldb,
                       void* __restrict__ Cv, int ldc,
                       const float* __restrict__ bias, int nbias,
                       const float* __restrict__ wnorm,
                       long M, int N, int K,
                       long arpb, long btstride, long biasstride) {
  int lane = threadIdx.x;
  int tilesN = N >> 6;
  long tiles = (M >> 5) * (long)tilesN;
  long wid = (long)blockIdx.x * 8 + threadIdx.y;
  if (wid >= tiles) return;
  long tm = wid / tilesN;
  int  tn = (int)(wid % tilesN);
  long row0 = tm << 5;
  int  col0 = tn << 6;
  int  blk  = (int)(row0 / arpb);
  const _Float16* Bp = BT + (long)blk * btstride;

  v8f acc[2][4] = { { vzero8(), vzero8(), vzero8(), vzero8() },
                    { vzero8(), vzero8(), vzero8(), vzero8() } };
  for (int k0 = 0; k0 < K; k0 += 32) {
    if (k0 + 32 < K) {
      __builtin_prefetch(A + (row0 + (lane & 15)) * (long)lda + k0 + 32, 0, 3);
      __builtin_prefetch(A + (row0 + 16 + (lane & 15)) * (long)lda + k0 + 32, 0, 3);
    }
    v16h a0 = frag_a_g(A, row0,      lda, k0);
    v16h a1 = frag_a_g(A, row0 + 16, lda, k0);
#pragma unroll
    for (int f = 0; f < 4; ++f) {
      v16h b = frag_b_g(Bp, col0 + f * 16, ldb, k0);
      acc[0][f] = __builtin_amdgcn_wmma_f32_16x16x32_f16(false, a0, false, b,
                                                         (short)0, acc[0][f], false, false);
      acc[1][f] = __builtin_amdgcn_wmma_f32_16x16x32_f16(false, a1, false, b,
                                                         (short)0, acc[1][f], false, false);
    }
  }
  int g = lane >> 4, c = lane & 15;

  float scale[2][8];
  if (EPI == 2) {   // RMSNorm over the 64 cols of each 16-row tile (== one head)
#pragma unroll
    for (int rb = 0; rb < 2; ++rb) {
#pragma unroll
      for (int v = 0; v < 8; ++v) {
        float s2 = 0.f;
#pragma unroll
        for (int f = 0; f < 4; ++f) s2 += acc[rb][f][v] * acc[rb][f][v];
        scale[rb][v] = s2;
      }
#pragma unroll
      for (int m = 1; m <= 8; m <<= 1)
#pragma unroll
        for (int v = 0; v < 8; ++v) scale[rb][v] += __shfl_xor(scale[rb][v], m, 32);
#pragma unroll
      for (int v = 0; v < 8; ++v) scale[rb][v] = rsqrtf(scale[rb][v] * (1.f / 64.f) + EPSR);
    }
  }

  const float* bp = HASBIAS ? (bias + (long)blk * biasstride) : nullptr;
#pragma unroll
  for (int f = 0; f < 4; ++f) {
    int col = col0 + f * 16 + c;
    float bv = 0.f;
    if (HASBIAS) bv = (col < nbias) ? bp[col] : 0.f;
#pragma unroll
    for (int rb = 0; rb < 2; ++rb) {
#pragma unroll
      for (int v = 0; v < 8; ++v) {
        long row = row0 + rb * 16 + v + 8 * g;
        float val = acc[rb][f][v] + bv;
        if (EPI == 2) val = acc[rb][f][v] * scale[rb][v] * wnorm[f * 16 + c];
        if (EPI == 0) ((float*)Cv)[row * (long)ldc + col] = val;
        else          ((_Float16*)Cv)[row * (long)ldc + col] = (_Float16)val;
      }
    }
  }
}

// ---------------------------------------------------------------------------
// Ensemble self-attention: one wave per (l,b,h, 16-query tile).  N=512, dh=64.
// S = Q K^T (WMMA), fp32 softmax in LDS (no 1/sqrt(dh) scale: faithful to
// reference), O = P V (WMMA, P converted f32->f16 while building fragments).
// ---------------------------------------------------------------------------
__global__ void k_attn(const _Float16* __restrict__ q16, const _Float16* __restrict__ kv16,
                       const _Float16* __restrict__ vt16, float* __restrict__ obuf) {
  extern __shared__ float smem[];           // 16 rows x 512 f32 scores = 32KB
  int lane = threadIdx.x;
  int wid = blockIdx.x;
  int qt = wid & 31;
  int h  = (wid >> 5) & 7;
  int b  = (wid >> 8) & 3;
  int l  = wid >> 10;
  long rowbase = ((long)l * 4 + b) * 512;
  long n0 = rowbase + qt * 16;

  v16h qa0 = frag_a_g(q16, n0, 512, h * 64);
  v16h qa1 = frag_a_g(q16, n0, 512, h * 64 + 32);
  int g = lane >> 4, c = lane & 15;

  for (int j0 = 0; j0 < 512; j0 += 16) {   // K rows act as B^T rows directly
    v16h kb0 = frag_b_g(kv16, rowbase + j0, 1024, h * 64);
    v16h kb1 = frag_b_g(kv16, rowbase + j0, 1024, h * 64 + 32);
    v8f s = vzero8();
    s = __builtin_amdgcn_wmma_f32_16x16x32_f16(false, qa0, false, kb0, (short)0, s, false, false);
    s = __builtin_amdgcn_wmma_f32_16x16x32_f16(false, qa1, false, kb1, (short)0, s, false, false);
#pragma unroll
    for (int v = 0; v < 8; ++v) smem[(v + 8 * g) * 512 + j0 + c] = s[v];
  }
  __syncthreads();
  { // fp32 softmax: 2 lanes per row, 256 elems each
    int r = lane & 15, hf = lane >> 4;
    float* rp = smem + r * 512 + hf * 256;
    float mx = -3.0e38f;
    for (int j = 0; j < 256; ++j) mx = fmaxf(mx, rp[j]);
    mx = fmaxf(mx, __shfl_xor(mx, 16, 32));
    float sum = 0.f;
    for (int j = 0; j < 256; ++j) { float e = __expf(rp[j] - mx); rp[j] = e; sum += e; }
    sum += __shfl_xor(sum, 16, 32);
    float inv = 1.f / sum;
    for (int j = 0; j < 256; ++j) rp[j] *= inv;
  }
  __syncthreads();

  long vtbase = (((long)l * 4 + b) * 8 + h) * 64;     // vt16: (l,b,h,dh,n)
  v8f o[4] = { vzero8(), vzero8(), vzero8(), vzero8() };
  for (int j0 = 0; j0 < 512; j0 += 32) {
    v16h pa = frag_a_lds_f32(smem, 512, j0);
#pragma unroll
    for (int f = 0; f < 4; ++f) {
      v16h vb = frag_b_g(vt16, vtbase + f * 16, 512, j0);
      o[f] = __builtin_amdgcn_wmma_f32_16x16x32_f16(false, pa, false, vb, (short)0, o[f], false, false);
    }
  }
  long obase = ((((long)l * 4 + b) * 8 + h) * 512) * 64;  // obuf: (l,b,h,n,dh)
#pragma unroll
  for (int f = 0; f < 4; ++f)
#pragma unroll
    for (int v = 0; v < 8; ++v)
      obuf[obase + (long)(qt * 16 + v + 8 * g) * 64 + f * 16 + c] = o[f][v];
}

// ---------------------------------------------------------------------------
// Pooled cross-message attention: <=36 keys/token -> VALU path, one wave per
// token; lane = 4*head + quarter of dh. Writes gated merged heads (f16).
// ---------------------------------------------------------------------------
__global__ void k_pooled(const _Float16* __restrict__ q16, const _Float16* __restrict__ kbuf,
                         const _Float16* __restrict__ vbuf, const float* __restrict__ gates,
                         _Float16* __restrict__ out16, int M, int bcast,
                         const _Float16* __restrict__ qvec, const float* __restrict__ qg) {
  __shared__ float sc[8 * 40];
  int lane = threadIdx.x;
  int h = lane >> 2, p = lane & 3;
  long qrow = blockIdx.x;
  int n = (int)(qrow & 511);
  int b = bcast ? (int)(qrow >> 9) : (int)((qrow >> 9) & 3);

  const _Float16* qp = (bcast ? qvec : (q16 + qrow * 512)) + h * 64 + p * 16;
  float qr[16];
#pragma unroll
  for (int i = 0; i < 16; ++i) qr[i] = (float)qp[i];

  float mx = -3.0e38f;
  for (int j = 0; j < M; ++j) {            // key row: ((s*L+l2)*B+b)*N+n == (j*4+b)*512+n
    const _Float16* kp = kbuf + (((long)j * 4 + b) * 512 + n) * 512 + h * 64 + p * 16;
    float d = 0.f;
#pragma unroll
    for (int i = 0; i < 16; ++i) d += qr[i] * (float)kp[i];
    d += __shfl_xor(d, 1, 32);
    d += __shfl_xor(d, 2, 32);
    if (p == 0) sc[h * 40 + j] = d;
    mx = fmaxf(mx, d);
  }
  __syncthreads();
  float sum = 0.f;
  for (int j = p; j < M; j += 4) { float e = __expf(sc[h * 40 + j] - mx); sc[h * 40 + j] = e; sum += e; }
  __syncthreads();
  sum += __shfl_xor(sum, 1, 32);
  sum += __shfl_xor(sum, 2, 32);
  float inv = 1.f / sum;

  float o[16];
#pragma unroll
  for (int i = 0; i < 16; ++i) o[i] = 0.f;
  for (int j = 0; j < M; ++j) {
    float w = sc[h * 40 + j] * inv;
    const _Float16* vp = vbuf + (((long)j * 4 + b) * 512 + n) * 512 + h * 64 + p * 16;
#pragma unroll
    for (int i = 0; i < 16; ++i) o[i] += w * (float)vp[i];
  }
  float gt = bcast ? qg[h] : gates[qrow * 8 + h];
  _Float16* op = out16 + qrow * 512 + h * 64 + p * 16;
#pragma unroll
  for (int i = 0; i < 16; ++i) op[i] = (_Float16)(o[i] * gt);
}

// ----------------------------- elementwise helpers -------------------------

__global__ void k_rmsnorm(const float* __restrict__ x, const float* __restrict__ w,
                          long wstride, long rpb, _Float16* __restrict__ y, long rows) {
  long row = (long)blockIdx.x * blockDim.y + threadIdx.y;
  if (row >= rows) return;
  int lane = threadIdx.x;
  const float* xp = x + row * 512 + lane * 16;
  float v[16]; float ss = 0.f;
#pragma unroll
  for (int i = 0; i < 16; ++i) { v[i] = xp[i]; ss += v[i] * v[i]; }
#pragma unroll
  for (int m = 1; m <= 16; m <<= 1) ss += __shfl_xor(ss, m, 32);
  float sc = rsqrtf(ss * (1.f / 512.f) + EPSR);
  const float* wp = w + (row / rpb) * wstride + lane * 16;
  _Float16* yp = y + row * 512 + lane * 16;
#pragma unroll
  for (int i = 0; i < 16; ++i) yp[i] = (_Float16)(v[i] * sc * wp[i]);
}

__global__ void k_gates(const _Float16* __restrict__ t, const float* __restrict__ wg,
                        long wgstride, long rpb, float* __restrict__ out, long rows) {
  long row = (long)blockIdx.x * blockDim.y + threadIdx.y;
  if (row >= rows) return;
  int lane = threadIdx.x;
  const _Float16* tp = t + row * 512 + lane * 16;
  const float* w = wg + (row / rpb) * wgstride;
  float acc[8] = {0,0,0,0,0,0,0,0};
#pragma unroll
  for (int i = 0; i < 16; ++i) {
    float tv = (float)tp[i];
    const float* wr = w + (lane * 16 + i) * 8;
#pragma unroll
    for (int hh = 0; hh < 8; ++hh) acc[hh] += tv * wr[hh];
  }
#pragma unroll
  for (int m = 1; m <= 16; m <<= 1)
#pragma unroll
    for (int hh = 0; hh < 8; ++hh) acc[hh] += __shfl_xor(acc[hh], m, 32);
  if (lane < 8) out[row * 8 + lane] = 1.f / (1.f + __expf(-acc[lane]));
}

__global__ void k_wtrans(const float* __restrict__ in, _Float16* __restrict__ out,
                         int K, int Nn, int NP, int KP, long total) {
  long idx = (long)blockIdx.x * 256 + threadIdx.x;
  if (idx >= total) return;
  long per = (long)NP * KP;
  int blk = (int)(idx / per);
  long rem = idx % per;
  int nn = (int)(rem / KP), kk = (int)(rem % KP);
  float v = 0.f;
  if (nn < Nn && kk < K) v = in[((long)blk * K + kk) * Nn + nn];
  out[idx] = (_Float16)v;
}

__global__ void k_bcast(const float* __restrict__ tokens, float* __restrict__ tok,
                        _Float16* __restrict__ m0, long total) {
  long idx = (long)blockIdx.x * 256 + threadIdx.x;
  if (idx >= total) return;
  float v = tokens[idx & 1048575];   // B*N*D = 2^20
  tok[idx] = v;
  m0[idx] = (_Float16)v;
}

__global__ void k_vt(const _Float16* __restrict__ kv16, _Float16* __restrict__ vt, long total) {
  long idx = (long)blockIdx.x * 256 + threadIdx.x;
  if (idx >= total) return;
  int n = (int)(idx & 511), d = (int)((idx >> 9) & 63);
  int h = (int)((idx >> 15) & 7), b = (int)((idx >> 18) & 3), l = (int)(idx >> 20);
  vt[idx] = kv16[((((long)l * 4 + b) * 512 + n) * 1024) + 512 + h * 64 + d];
}

__global__ void k_merge_gate(const float* __restrict__ obuf, const float* __restrict__ gates,
                             _Float16* __restrict__ am, long total) {
  long idx = (long)blockIdx.x * 256 + threadIdx.x;
  if (idx >= total) return;
  long row = idx >> 9; int col = (int)(idx & 511);
  int h = col >> 6, d = col & 63;
  int n = (int)(row & 511), b = (int)((row >> 9) & 3), l = (int)(row >> 11);
  float o = obuf[((((long)l * 4 + b) * 8 + h) * 512 + n) * 64 + d];
  am[idx] = (_Float16)(o * gates[row * 8 + h]);
}

__global__ void k_geglu(const float* __restrict__ hff, _Float16* __restrict__ g16, long total) {
  long idx = (long)blockIdx.x * 256 + threadIdx.x;
  if (idx >= total) return;
  long row = idx / 1376; int j = (int)(idx % 1376);
  float r = 0.f;
  if (j < 1365) {
    float sim = hff[row * 2752 + j];
    float gt  = hff[row * 2752 + 1365 + j];
    r = sim * 0.5f * gt * (1.f + erff(gt * 0.70710678118654752f));  // exact GELU
  }
  g16[idx] = (_Float16)r;
}

__global__ void k_finalq(const float* __restrict__ qr, const float* __restrict__ rnorm,
                         const float* __restrict__ wq, const float* __restrict__ wg,
                         _Float16* __restrict__ qvec, float* __restrict__ qg) {
  int e = threadIdx.x;
  float ss = 0.f;
  for (int d = 0; d < 512; ++d) ss += qr[d] * qr[d];
  float sc = rsqrtf(ss * (1.f / 512.f) + EPSR);
  float acc = 0.f;
  for (int d = 0; d < 512; ++d) acc += qr[d] * sc * rnorm[d] * wq[(long)d * 512 + e];
  qvec[e] = (_Float16)acc;
  if (e < 8) {
    float a = 0.f;
    for (int d = 0; d < 512; ++d) a += qr[d] * sc * rnorm[d] * wg[d * 8 + e];
    qg[e] = 1.f / (1.f + __expf(-a));
  }
}

// ----------------------------- host side -----------------------------------

static void run_gemm(int epi, bool hasbias, const _Float16* A, int lda,
                     const _Float16* BT, int ldb, void* C, int ldc,
                     const float* bias, int nbias, const float* wnorm,
                     long M, int N, int K, long arpb, long btstride, long biasstride,
                     hipStream_t s) {
  long tiles = (M >> 5) * (long)(N >> 6);
  dim3 blk(32, 8, 1);
  unsigned grid = (unsigned)((tiles + 7) >> 3);
  if (epi == 0 && !hasbias)
    k_gemm<0,0><<<grid, blk, 0, s>>>(A, lda, BT, ldb, C, ldc, bias, nbias, wnorm, M, N, K, arpb, btstride, biasstride);
  else if (epi == 0)
    k_gemm<0,1><<<grid, blk, 0, s>>>(A, lda, BT, ldb, C, ldc, bias, nbias, wnorm, M, N, K, arpb, btstride, biasstride);
  else if (epi == 1 && !hasbias)
    k_gemm<1,0><<<grid, blk, 0, s>>>(A, lda, BT, ldb, C, ldc, bias, nbias, wnorm, M, N, K, arpb, btstride, biasstride);
  else if (epi == 1)
    k_gemm<1,1><<<grid, blk, 0, s>>>(A, lda, BT, ldb, C, ldc, bias, nbias, wnorm, M, N, K, arpb, btstride, biasstride);
  else
    k_gemm<2,0><<<grid, blk, 0, s>>>(A, lda, BT, ldb, C, ldc, bias, nbias, wnorm, M, N, K, arpb, btstride, biasstride);
}

extern "C" void kernel_launch(void* const* d_in, const int* in_sizes, int n_in,
                              void* d_out, int out_size, void* d_ws, size_t ws_size,
                              hipStream_t stream) {
  (void)in_sizes; (void)n_in; (void)out_size; (void)ws_size;
  const float* tokens   = (const float*)d_in[0];
  const float* ea_norm  = (const float*)d_in[1];
  const float* ea_wq    = (const float*)d_in[2];
  const float* ea_wkv   = (const float*)d_in[3];
  const float* ea_wo    = (const float*)d_in[4];
  const float* ea_wg    = (const float*)d_in[5];
  const float* ef_norm  = (const float*)d_in[6];
  const float* ef_wk    = (const float*)d_in[7];
  const float* ef_bk    = (const float*)d_in[8];
  const float* ef_wv    = (const float*)d_in[9];
  const float* ef_bv    = (const float*)d_in[10];
  const float* ra_norm  = (const float*)d_in[11];
  const float* ra_knorm = (const float*)d_in[12];
  const float* ra_wq    = (const float*)d_in[13];
  const float* ra_wkv   = (const float*)d_in[14];
  const float* ra_wo    = (const float*)d_in[15];
  const float* ra_wg    = (const float*)d_in[16];
  const float* q_read   = (const float*)d_in[17];

  const long R = 8192;   // L*B*N token rows per ensemble pass
  char* wsp = (char*)d_ws;
  auto alloc = [&](size_t elems, size_t esz) -> char* {
    char* p = wsp;
    wsp += (elems * esz + 255) & ~(size_t)255;
    return p;
  };
  _Float16* msgs   = (_Float16*)alloc(9L * R * 512, 2);   // message cache (f16)
  float*    tok    = (float*)   alloc(R * 512, 4);
  _Float16* t16    = (_Float16*)alloc(R * 512, 2);
  _Float16* q16    = (_Float16*)alloc(R * 512, 2);
  _Float16* kv16   = (_Float16*)alloc(R * 1024, 2);
  _Float16* vt16   = (_Float16*)alloc(R * 512, 2);
  float*    ag     = (float*)   alloc(R * 8, 4);
  float*    obuf   = (float*)   alloc(R * 512, 4);
  _Float16* am16   = (_Float16*)alloc(R * 512, 2);
  float*    hff    = (float*)   alloc(R * 2752, 4);
  _Float16* g16    = (_Float16*)alloc(R * 1376, 2);
  _Float16* kbuf   = (_Float16*)alloc(9L * R * 512, 2);   // incremental ctx K (knorm'd)
  _Float16* vbuf   = (_Float16*)alloc(9L * R * 512, 2);   // incremental ctx V
  _Float16* rm16   = (_Float16*)alloc(2048L * 512, 2);
  _Float16* qvec   = (_Float16*)alloc(512, 2);
  float*    qg     = (float*)   alloc(8, 4);
  _Float16* w_eaq  = (_Float16*)alloc(4L * 512 * 512, 2);
  _Float16* w_eakv = (_Float16*)alloc(4L * 1024 * 512, 2);
  _Float16* w_eao  = (_Float16*)alloc(4L * 512 * 512, 2);
  _Float16* w_efk  = (_Float16*)alloc(4L * 2752 * 512, 2);
  _Float16* w_efv  = (_Float16*)alloc(4L * 512 * 1376, 2);
  _Float16* w_raq  = (_Float16*)alloc(512L * 512, 2);
  _Float16* w_rakv = (_Float16*)alloc(1024L * 512, 2);
  _Float16* w_rao  = (_Float16*)alloc(512L * 512, 2);

  auto gE = [](long total) { return (unsigned)((total + 255) / 256); };

  // weight transpose+cast to f16 B^T (N x K), padded to 64/32 multiples
  k_wtrans<<<gE(4L*512*512),  256, 0, stream>>>(ea_wq,  w_eaq,  512, 512, 512, 512,  4L*512*512);
  k_wtrans<<<gE(4L*1024*512), 256, 0, stream>>>(ea_wkv, w_eakv, 512, 1024, 1024, 512, 4L*1024*512);
  k_wtrans<<<gE(4L*512*512),  256, 0, stream>>>(ea_wo,  w_eao,  512, 512, 512, 512,  4L*512*512);
  k_wtrans<<<gE(4L*2752*512), 256, 0, stream>>>(ef_wk,  w_efk,  512, 2730, 2752, 512, 4L*2752*512);
  k_wtrans<<<gE(4L*512*1376), 256, 0, stream>>>(ef_wv,  w_efv,  1365, 512, 512, 1376, 4L*512*1376);
  k_wtrans<<<gE(512L*512),    256, 0, stream>>>(ra_wq,  w_raq,  512, 512, 512, 512,  512L*512);
  k_wtrans<<<gE(1024L*512),   256, 0, stream>>>(ra_wkv, w_rakv, 512, 1024, 1024, 512, 1024L*512);
  k_wtrans<<<gE(512L*512),    256, 0, stream>>>(ra_wo,  w_rao,  512, 512, 512, 512,  512L*512);

  k_bcast<<<gE(R*512), 256, 0, stream>>>(tokens, tok, msgs, R * 512);  // msgs slot 0

  dim3 rblk(32, 8, 1);
  for (int step = 0; step < 4; ++step) {
    // ---- ensemble attention ----
    k_rmsnorm<<<(unsigned)((R + 7) / 8), rblk, 0, stream>>>(tok, ea_norm, 512, 2048, t16, R);
    run_gemm(1, false, t16, 512, w_eaq, 512, q16, 512, nullptr, 0, nullptr,
             R, 512, 512, 2048, 512L*512, 0, stream);
    run_gemm(1, false, t16, 512, w_eakv, 512, kv16, 1024, nullptr, 0, nullptr,
             R, 1024, 512, 2048, 1024L*512, 0, stream);
    k_vt<<<gE(R*512), 256, 0, stream>>>(kv16, vt16, R * 512);
    k_gates<<<(unsigned)((R + 7) / 8), rblk, 0, stream>>>(t16, ea_wg, 512L*8, 2048, ag, R);
    k_attn<<<4096, 32, 16 * 512 * 4, stream>>>(q16, kv16, vt16, obuf);
    k_merge_gate<<<gE(R*512), 256, 0, stream>>>(obuf, ag, am16, R * 512);
    int slotA = 2 * step + 1;
    run_gemm(1, false, am16, 512, w_eao, 512, msgs + (long)slotA * R * 512, 512,
             nullptr, 0, nullptr, R, 512, 512, 2048, 512L*512, 0, stream);
    // ---- ensemble GEGLU FF ----
    k_rmsnorm<<<(unsigned)((R + 7) / 8), rblk, 0, stream>>>(tok, ef_norm, 512, 2048, t16, R);
    run_gemm(0, true, t16, 512, w_efk, 512, hff, 2752, ef_bk, 2730, nullptr,
             R, 2752, 512, 2048, 2752L*512, 2730, stream);
    k_geglu<<<gE(R*1376), 256, 0, stream>>>(hff, g16, R * 1376);
    int slotF = 2 * step + 2;
    run_gemm(1, true, g16, 1376, w_efv, 1376, msgs + (long)slotF * R * 512, 512,
             ef_bv, 512, nullptr, R, 512, 1376, 2048, 512L*1376, 512, stream);
    // ---- project only the NEW message slots to context K/V (incremental) ----
    int s0 = (step == 0) ? 0 : slotA;
    long Mp = (long)((step == 0) ? 3 : 2) * R;
    run_gemm(2, false, msgs + (long)s0 * R * 512, 512, w_rakv, 512,
             kbuf + (long)s0 * R * 512, 512, nullptr, 0, ra_knorm,
             Mp, 512, 512, Mp, 0, 0, stream);                 // fused key-RMSNorm
    run_gemm(1, false, msgs + (long)s0 * R * 512, 512, w_rakv + 512L*512, 512,
             vbuf + (long)s0 * R * 512, 512, nullptr, 0, nullptr,
             Mp, 512, 512, Mp, 0, 0, stream);
    // ---- pooled message-exchange attention updates tok ----
    if (step < 3) {
      int Mctx = (2 * step + 3) * 4;
      k_rmsnorm<<<(unsigned)((R + 7) / 8), rblk, 0, stream>>>(tok, ra_norm, 0, 1, t16, R);
      run_gemm(1, false, t16, 512, w_raq, 512, q16, 512, nullptr, 0, nullptr,
               R, 512, 512, R, 0, 0, stream);
      k_gates<<<(unsigned)((R + 7) / 8), rblk, 0, stream>>>(t16, ra_wg, 0, 1, ag, R);
      k_pooled<<<(unsigned)R, 32, 0, stream>>>(q16, kbuf, vbuf, ag, am16, Mctx, 0, nullptr, nullptr);
      run_gemm(0, false, am16, 512, w_rao, 512, tok, 512, nullptr, 0, nullptr,
               R, 512, 512, R, 0, 0, stream);
    }
  }
  // ---- final pooled readout with broadcast query over all 36 messages ----
  k_finalq<<<1, 512, 0, stream>>>(q_read, ra_norm, ra_wq, ra_wg, qvec, qg);
  k_pooled<<<2048, 32, 0, stream>>>(nullptr, kbuf, vbuf, nullptr, rm16, 36, 1, qvec, qg);
  run_gemm(0, false, rm16, 512, w_rao, 512, (float*)d_out, 512, nullptr, 0, nullptr,
           2048, 512, 512, 2048, 0, 0, stream);
}
